// GraphRNNBilinear_69097433858699
// MI455X (gfx1250) — matrix operand order repaired and checked by reading, not compiled, not run on hardware
//
#include <hip/hip_runtime.h>

// GraphRNNBilinear for MI455X (gfx1250): bf16 WMMA pipeline.
// Stages: weight cvt (N/K padded) -> build x -> xproj GEMMs ->
//         LDS-resident RNN (fwd+bwd) -> proj/head/dep GEMMs (ELU fused) ->
//         WMMA estep scan -> band gather.
// GEMM inner loops are branchless (weights padded to tile coverage) so WMMA
// always executes with full EXEC; only epilogue stores are guarded.

typedef __attribute__((ext_vector_type(8)))  float  v8f;
typedef __attribute__((ext_vector_type(8)))  __bf16 v8bf;
typedef __attribute__((ext_vector_type(16))) __bf16 v16bf;

#define B_   64
#define T_   383
#define ENC_ 768
#define TAG_ 128
#define HG_  400
#define HGP_ 416   // HG padded to multiple of 32 (13 K-slices)
#define HGN_ 512   // HG padded to GEMM BN coverage (4 * 128)
#define HE_  256
#define M_   20
#define D_   896
#define S_   384
#define BS_  (B_*S_)   // 24576

static __device__ __forceinline__ v16bf cat16(v8bf lo, v8bf hi){
  v16bf r;
#pragma unroll
  for(int i=0;i<8;i++){ r[i]=lo[i]; r[i+8]=hi[i]; }
  return r;
}
static __device__ __forceinline__ v8f wmma_bf16(v16bf a, v16bf b, v8f c){
  return __builtin_amdgcn_wmma_f32_16x16x32_bf16(false, a, false, b, (short)0, c, false, false);
}
// A fragment: 16x32 bf16 tile, p -> element (row0,k0), row stride ld.
// lane l: row=l%16; g=l/16; k = e+8g (e<8), 16+(e-8)+8g (e>=8) -> two 16B loads.
static __device__ __forceinline__ v16bf load_a(const __bf16* p, int ld, int lane){
  int r = lane & 15, g = lane >> 4;
  const __bf16* q = p + r*ld + 8*g;
  return cat16(*(const v8bf*)q, *(const v8bf*)(q+16));
}
// B fragment: 32x16, sourced from W(N,K) row-major (computes A@W^T).
// lane l: n=l%16; k = e+8g (e<8), 16+(e-8)+8g (e>=8) -> two 16B loads.
static __device__ __forceinline__ v16bf load_b(const __bf16* p, int ld, int lane){
  int r = lane & 15, g = lane >> 4;
  const __bf16* q = p + r*ld + 8*g;
  return cat16(*(const v8bf*)q, *(const v8bf*)(q+16));
}

// ---- weight f32 (N,K) -> bf16 (Np,Kp) zero padded --------------------------
__global__ void k_cvt(const float* __restrict__ W, __bf16* __restrict__ O,
                      int N, int K, int Np, int Kp){
  int i = blockIdx.x*256 + threadIdx.x;
  if(i >= Np*Kp) return;
  int n = i / Kp, k = i % Kp;
  O[i] = (__bf16)((k < K && n < N) ? W[n*K + k] : 0.f);
}

// ---- build x = [sentinel ; input||tag] as bf16 (BS x D) --------------------
__global__ void k_build_x(const float* __restrict__ inp, const float* __restrict__ tag,
                          const float* __restrict__ sent, __bf16* __restrict__ xbf){
  long i = (long)blockIdx.x*256 + threadIdx.x;
  if(i >= (long)BS_*D_) return;
  int d = (int)(i % D_); long r = i / D_;
  int b = (int)(r / S_), s = (int)(r % S_);
  float v;
  if(s == 0)          v = sent[d];
  else if(d < ENC_)   v = inp[((long)b*T_ + (s-1))*ENC_ + d];
  else                v = tag[((long)b*T_ + (s-1))*TAG_ + (d-ENC_)];
  xbf[i] = (__bf16)v;
}

// ---- generic C(M,N) = A(M,K)bf16 @ W(Np,K)bf16^T + bias, opt. ELU ----------
// block: 256 thr (8 waves). BM=64, BN=128. wave w: mtile=w%4, 4 ntiles.
// W must be row-padded so every tile covered by the grid is loadable; only the
// stores are guarded (keeps EXEC all-ones through every WMMA).
__global__ __launch_bounds__(256)
void k_gemm(const __bf16* __restrict__ A, int lda,
            const __bf16* __restrict__ W, int ldw,
            const float* __restrict__ bias,
            float* __restrict__ C, int ldc, int N, int K, int act){
  int tid = threadIdx.x, lane = tid & 31, w = tid >> 5;
  int mt = w & 3, ntb = (w >> 2) * 4;
  int rowbase = blockIdx.x*64 + mt*16;
  int col00   = (blockIdx.y*8 + ntb)*16;
  int r = lane & 15, g = lane >> 4;
  v8f acc[4] = {};
  const __bf16* Abase = A + (long)rowbase*lda;
  const __bf16* Wb    = W + (long)col00*ldw;
  int nk = K >> 5;
  for(int ks=0; ks<nk; ks++){
    v16bf af = load_a(Abase + ks*32, lda, lane);
    if(ks + 2 < nk)
      __builtin_prefetch(Abase + (ks+2)*32 + (lane&15)*lda, 0, 3);
#pragma unroll
    for(int j=0;j<4;j++){
      v16bf bf = load_b(Wb + (long)j*16*ldw + ks*32, ldw, lane);
      acc[j] = wmma_bf16(af, bf, acc[j]);
    }
  }
#pragma unroll
  for(int j=0;j<4;j++){
    int col = col00 + j*16 + r;
    if(col < N){
      float bb = bias[col];
#pragma unroll
      for(int i=0;i<8;i++){
        float v = acc[j][i] + bb;
        if(act == 1) v = (v > 0.f) ? v : expm1f(v);   // ELU(alpha=1)
        C[(long)(rowbase + i + 8*g)*ldc + col] = v;
      }
    }
  }
}

// ---- masked RNN recurrence, h LDS-resident, fwd(blk0)/bwd(blk1) ------------
// block: 800 thr (25 waves). wave w owns N-tile w (cols 16w..16w+15), 4 M-tiles.
__global__ __launch_bounds__(800)
void k_rnn(const float* __restrict__ xpF, const float* __restrict__ xpB,
           const __bf16* __restrict__ WhhF, const __bf16* __restrict__ WhhB,
           const float* __restrict__ bhhF, const float* __restrict__ bhhB,
           const int* __restrict__ mask, __bf16* __restrict__ gs){
  extern __shared__ __bf16 lds[];           // 2 * 64 * HGP_
  int dir = blockIdx.x;
  const float*  xproj = dir ? xpB  : xpF;
  const __bf16* Whh   = dir ? WhhB : WhhF;
  const float*  bhh   = dir ? bhhB : bhhF;
  __bf16* h0 = lds;
  __bf16* h1 = lds + 64*HGP_;
  int tid = threadIdx.x, lane = tid & 31, w = tid >> 5;
  for(int i=tid; i<2*64*HGP_; i+=800) lds[i] = (__bf16)0.f;   // zero h + pad cols
  __syncthreads();
  int r = lane & 15, g = lane >> 4;
  int col = w*16 + r;
  float bb = bhh[col];
  const __bf16* Bb = Whh + (long)(w*16)*HGP_;
  for(int step=0; step<S_; step++){
    int t = dir ? (S_-1-step) : step;
    __bf16* hc = (step & 1) ? h1 : h0;
    __bf16* hn = (step & 1) ? h0 : h1;
#pragma unroll
    for(int mtl=0; mtl<4; mtl++){
      v8f acc = {};
      const __bf16* Ab = hc + mtl*16*HGP_;
#pragma unroll 4
      for(int ks=0; ks<13; ks++){
        v16bf af = load_a(Ab + ks*32, HGP_, lane);
        v16bf bf = load_b(Bb + ks*32, HGP_, lane);
        acc = wmma_bf16(af, bf, acc);
      }
#pragma unroll
      for(int i=0;i<8;i++){
        int b = mtl*16 + i + 8*g;                     // batch row
        float pre = acc[i] + xproj[((long)b*S_ + t)*HG_ + col] + bb;
        float hv  = tanhf(pre);
        float mf  = (t == 0) ? 1.f : (float)mask[b*T_ + (t-1)];
        float hold = (float)hc[b*HGP_ + col];
        float hnew = mf*hv + (1.f - mf)*hold;
        hn[b*HGP_ + col] = (__bf16)hnew;
        gs[((long)b*S_ + t)*(2*HG_) + dir*HG_ + col] = (__bf16)(hnew*mf);
      }
    }
    __syncthreads();
  }
}

// ---- estep: 20-step scan, 16 independent rows per block, WMMA 16x256x256 ---
// block: 128 thr (4 waves); wave w owns N-tiles 4w..4w+3.
__global__ __launch_bounds__(128)
void k_estep(const float* __restrict__ proj, const __bf16* __restrict__ WhhE,
             const float* __restrict__ wihE, const float* __restrict__ bihE,
             const float* __restrict__ bhhE, const __bf16* __restrict__ WclsBf,
             const float* __restrict__ bcls, const float* __restrict__ bos,
             float* __restrict__ Ap){
  __shared__ __bf16 hb[2][16][HE_];
  __shared__ float  xe[16];
  int tid = threadIdx.x, lane = tid & 31, w = tid >> 5;
  long rowBase = (long)blockIdx.x*16;
  for(int i=tid; i<16*HE_; i+=128){
    int rr = i / HE_, cc = i % HE_;
    hb[0][rr][cc] = (__bf16)proj[(rowBase+rr)*HE_ + cc];
  }
  if(tid < 16) xe[tid] = bos[0];
  __syncthreads();
  int r = lane & 15, g = lane >> 4;
  for(int step=0; step<M_; step++){
    int cur = step & 1, nxt = cur ^ 1;
#pragma unroll
    for(int j=0;j<4;j++){
      int c0 = (w*4 + j)*16;
      v8f acc = {};
#pragma unroll
      for(int ks=0; ks<8; ks++){
        v16bf af = load_a(&hb[cur][0][0] + ks*32, HE_, lane);
        v16bf bf = load_b(WhhE + (long)c0*HE_ + ks*32, HE_, lane);
        acc = wmma_bf16(af, bf, acc);
      }
      int col = c0 + r;
      float be = bihE[col] + bhhE[col];
      float we = wihE[col];
#pragma unroll
      for(int i=0;i<8;i++){
        int row = i + 8*g;
        float pre = acc[i] + xe[row]*we + be;
        hb[nxt][row][col] = (__bf16)tanhf(pre);
      }
    }
    __syncthreads();
    if(tid < 16){
      float s = bcls[0];
      for(int n=0;n<HE_;n++) s += (float)hb[nxt][tid][n]*(float)WclsBf[n];
      Ap[(rowBase + tid)*M_ + step] = s;
      xe[tid] = s;
    }
    __syncthreads();
  }
}

// ---- banded gather into arc_logits (B,S,S) ---------------------------------
__global__ void k_band(const float* __restrict__ Ap, float* __restrict__ arc){
  long i = (long)blockIdx.x*256 + threadIdx.x;
  if(i >= (long)B_*S_*S_) return;
  int c = (int)(i % S_); long t = i / S_;
  int rr = (int)(t % S_); int b = (int)(t / S_);
  int start = (rr - M_ > 0) ? rr - M_ : 0;
  int k = c - start;
  bool valid = (c >= start) && (c < rr);
  int ik = k < 0 ? 0 : (k > M_-1 ? M_-1 : k);
  arc[i] = valid ? Ap[((long)b*S_ + rr)*M_ + ik] : 0.f;
}

extern "C" void kernel_launch(void* const* d_in, const int* in_sizes, int n_in,
                              void* d_out, int out_size, void* d_ws, size_t ws_size,
                              hipStream_t stream){
  const float* input  = (const float*)d_in[0];
  const float* tag    = (const float*)d_in[1];
  const int*   mask   = (const int*)  d_in[2];
  const float* sent   = (const float*)d_in[3];
  const float* WihF   = (const float*)d_in[4];
  const float* WhhF   = (const float*)d_in[5];
  const float* bihF   = (const float*)d_in[6];
  const float* bhhF   = (const float*)d_in[7];
  const float* WihB   = (const float*)d_in[8];
  const float* WhhB   = (const float*)d_in[9];
  const float* bihB   = (const float*)d_in[10];
  const float* bhhB   = (const float*)d_in[11];
  const float* Wg2e   = (const float*)d_in[12];
  const float* bg2e   = (const float*)d_in[13];
  const float* WihE   = (const float*)d_in[14];
  const float* WhhE   = (const float*)d_in[15];
  const float* bihE   = (const float*)d_in[16];
  const float* bhhE   = (const float*)d_in[17];
  const float* Wcls   = (const float*)d_in[18];
  const float* bcls   = (const float*)d_in[19];
  const float* bos    = (const float*)d_in[20];
  const float* Wht    = (const float*)d_in[21];
  const float* bht    = (const float*)d_in[22];
  const float* Wdt    = (const float*)d_in[23];
  const float* bdt    = (const float*)d_in[24];

  char* ws = (char*)d_ws;
  size_t off = 0;
  auto alloc = [&](size_t bytes)->char*{
    char* p = ws + off;
    off += (bytes + 255) & ~(size_t)255;
    return p;
  };
  __bf16* xbf    = (__bf16*)alloc((size_t)BS_*D_*2);
  __bf16* gs     = (__bf16*)alloc((size_t)BS_*2*HG_*2);
  float*  xpF    = (float*) alloc((size_t)BS_*HG_*4);
  float*  xpB    = (float*) alloc((size_t)BS_*HG_*4);
  float*  proj   = (float*) alloc((size_t)BS_*HE_*4);
  float*  Ap     = (float*) alloc((size_t)BS_*M_*4);
  __bf16* WihFb  = (__bf16*)alloc((size_t)HGN_*D_*2);      // N-padded 512
  __bf16* WihBb  = (__bf16*)alloc((size_t)HGN_*D_*2);
  __bf16* WhhFb  = (__bf16*)alloc((size_t)HG_*HGP_*2);     // K-padded 416
  __bf16* WhhBb  = (__bf16*)alloc((size_t)HG_*HGP_*2);
  __bf16* Wg2eb  = (__bf16*)alloc((size_t)HE_*2*HG_*2);
  __bf16* Whtb   = (__bf16*)alloc((size_t)TAG_*2*HG_*2);
  __bf16* Wdtb   = (__bf16*)alloc((size_t)TAG_*2*HG_*2);
  __bf16* WhhEb  = (__bf16*)alloc((size_t)HE_*HE_*2);
  __bf16* Wclsb  = (__bf16*)alloc((size_t)HE_*2);

  auto cvt = [&](const float* src, __bf16* dst, int N, int K, int Np, int Kp){
    int tot = Np*Kp;
    k_cvt<<<(tot+255)/256, 256, 0, stream>>>(src, dst, N, K, Np, Kp);
  };
  cvt(WihF, WihFb, HG_, D_, HGN_, D_);
  cvt(WihB, WihBb, HG_, D_, HGN_, D_);
  cvt(WhhF, WhhFb, HG_, HG_, HG_, HGP_);
  cvt(WhhB, WhhBb, HG_, HG_, HG_, HGP_);
  cvt(Wg2e, Wg2eb, HE_, 2*HG_, HE_, 2*HG_);
  cvt(Wht,  Whtb,  TAG_, 2*HG_, TAG_, 2*HG_);
  cvt(Wdt,  Wdtb,  TAG_, 2*HG_, TAG_, 2*HG_);
  cvt(WhhE, WhhEb, HE_, HE_, HE_, HE_);
  cvt(Wcls, Wclsb, 1, HE_, 1, HE_);

  {
    long tot = (long)BS_*D_;
    k_build_x<<<(int)((tot+255)/256), 256, 0, stream>>>(input, tag, sent, xbf);
  }

  dim3 blk(256);
  // input projections: (24576 x 896) @ (512pad x 896)^T, store-guarded to 400
  k_gemm<<<dim3(BS_/64, HGN_/128), blk, 0, stream>>>(
      xbf, D_, WihFb, D_, bihF, xpF, HG_, HG_, D_, 0);
  k_gemm<<<dim3(BS_/64, HGN_/128), blk, 0, stream>>>(
      xbf, D_, WihBb, D_, bihB, xpB, HG_, HG_, D_, 0);

  // recurrence (fwd + bwd in parallel), h ping-pong in dynamic LDS
  size_t rnnLds = (size_t)2*64*HGP_*sizeof(__bf16);
  k_rnn<<<2, 800, rnnLds, stream>>>(xpF, xpB, WhhFb, WhhBb, bhhF, bhhB, mask, gs);

  // graph_state projections (N = 256 / 128: exact tile coverage, no padding)
  k_gemm<<<dim3(BS_/64, HE_/128), blk, 0, stream>>>(
      gs, 2*HG_, Wg2eb, 2*HG_, bg2e, proj, HE_, HE_, 2*HG_, 0);

  float* out_head = (float*)d_out;
  float* out_dep  = out_head + (size_t)BS_*TAG_;
  float* out_arc  = out_dep  + (size_t)BS_*TAG_;
  k_gemm<<<dim3(BS_/64, TAG_/128), blk, 0, stream>>>(
      gs, 2*HG_, Whtb, 2*HG_, bht, out_head, TAG_, TAG_, 2*HG_, 1);
  k_gemm<<<dim3(BS_/64, TAG_/128), blk, 0, stream>>>(
      gs, 2*HG_, Wdtb, 2*HG_, bdt, out_dep, TAG_, TAG_, 2*HG_, 1);

  // estep scan -> Ap (BS x 20)
  k_estep<<<BS_/16, 128, 0, stream>>>(proj, WhhEb, WihE, bihE, bhhE,
                                      Wclsb, bcls, bos, Ap);

  // banded arc logits
  {
    long tot = (long)B_*S_*S_;
    k_band<<<(int)((tot+255)/256), 256, 0, stream>>>(Ap, out_arc);
  }
}
